// Model_33552284516756
// MI455X (gfx1250) — matrix operand order repaired and checked by reading, not compiled, not run on hardware
//
#include <hip/hip_runtime.h>
#include <math.h>

#define N_NODES 20000
#define F_IN    8710
#define HID     128
#define NCLS    70
#define NEDGE   640000
#define CPAD    80

typedef __attribute__((ext_vector_type(8)))  float          v8f;
typedef __attribute__((ext_vector_type(16))) __bf16         v16bf;
typedef __attribute__((ext_vector_type(8)))  unsigned short v8u16;

// ---- fp32 -> bf16 pair pack -------------------------------------------------
// Round each fp32 to nearest bf16 (ties-up) and pack the two high halves with a
// single v_perm_b32: result = {hi[31:16], lo[31:16]}.
#if __has_builtin(__builtin_amdgcn_perm)
__device__ __forceinline__ unsigned int f2bf_pk(float lo, float hi) {
  union { float f; unsigned int u; } a, b;
  a.f = lo; b.f = hi;
  return __builtin_amdgcn_perm(b.u + 0x8000u, a.u + 0x8000u, 0x07060302u);
}
#else
__device__ __forceinline__ unsigned int f2bf_pk(float lo, float hi) {
  union { float f; unsigned int u; } a, b;
  a.f = lo; b.f = hi;
  return ((a.u + 0x8000u) >> 16) | ((b.u + 0x8000u) & 0xFFFF0000u);
}
#endif

// A fragment: lanes 0-15 K=[kb..kb+7],[kb+16..kb+23] with kb = (lane>>4)*8
__device__ __forceinline__ v16bf frag_a(const unsigned short* base, int kb) {
  union { struct { v8u16 lo, hi; } s; v16bf v; } u;
  u.s.lo = *(const v8u16*)(base + kb);
  u.s.hi = *(const v8u16*)(base + kb + 16);
  return u.v;
}
// B fragment: K=[kB..kB+7],[kB+16..kB+23] with kB = (lane>>4)*8
__device__ __forceinline__ v16bf frag_b(const unsigned short* base, int kB) {
  union { struct { v8u16 lo, hi; } s; v16bf v; } u;
  u.s.lo = *(const v8u16*)(base + kB);
  u.s.hi = *(const v8u16*)(base + kB + 16);
  return u.v;
}

// ---------------- GEMM1: H1[N,128] = X[N,8710] @ W1[8710,128] (bf16 WMMA) ----
#define AST 40   // LDS row pitch in ushorts (80B: 16B aligned, 20-bank stride)

__device__ __forceinline__ void g1_fetch(const float* __restrict__ X,
                                         const float* __restrict__ W1,
                                         int m0, int k0, int t,
                                         float2 sa[4][2], float4 sb[2][2]) {
#pragma unroll
  for (int i = 0; i < 4; ++i) {          // A tile: 128 rows x 32 k
    int idx = t + (i << 8);
    int row = idx >> 3;
    int kq  = (idx & 7) << 2;
    int grow = m0 + row, gk = k0 + kq;
    float2 ab = make_float2(0.f, 0.f), cd = make_float2(0.f, 0.f);
    if (grow < N_NODES) {
      const float* p = X + (size_t)grow * F_IN + gk;   // 8-byte aligned (F_IN even)
      if (gk + 1 < F_IN) ab = *(const float2*)p;
      if (gk + 3 < F_IN) cd = *(const float2*)(p + 2);
      // warm L2 two K-tiles ahead on the dominant X stream (one line per row)
      if (((idx & 7) == 0) && (gk + 64 < F_IN))
        __builtin_prefetch(p + 64, 0, 1);
    }
    sa[i][0] = ab; sa[i][1] = cd;
  }
#pragma unroll
  for (int i = 0; i < 2; ++i) {          // B tile: 32 k x 128 n, k-pairs per thread
    int idx = t + (i << 8);              // 0..511
    int kk2 = idx >> 5;                  // 0..15 -> k = 2*kk2
    int nq  = (idx & 31) << 2;
    int gk  = k0 + (kk2 << 1);
    float4 w0 = make_float4(0.f, 0.f, 0.f, 0.f), w1 = w0;
    if (gk + 1 < F_IN) {                 // F_IN even: pairs never straddle the edge
      w0 = *(const float4*)(W1 + (size_t)gk * HID + nq);
      w1 = *(const float4*)(W1 + (size_t)(gk + 1) * HID + nq);
    }
    sb[i][0] = w0; sb[i][1] = w1;
  }
}

__device__ __forceinline__ void g1_stage(unsigned short* __restrict__ As,
                                         unsigned short* __restrict__ Bs,
                                         int t, const float2 sa[4][2],
                                         const float4 sb[2][2]) {
#pragma unroll
  for (int i = 0; i < 4; ++i) {          // A: packed dword stores, k-contiguous
    int idx = t + (i << 8);
    int row = idx >> 3;
    int kq  = (idx & 7) << 2;
    unsigned int* p = (unsigned int*)(As + row * AST + kq);   // kq%4==0 -> aligned
    p[0] = f2bf_pk(sa[i][0].x, sa[i][0].y);
    p[1] = f2bf_pk(sa[i][1].x, sa[i][1].y);
  }
#pragma unroll
  for (int i = 0; i < 2; ++i) {          // B transposed: Bs[n][k], k-pairs pack to dwords
    int idx = t + (i << 8);
    int kk2 = idx >> 5;
    int nq  = (idx & 31) << 2;
    float4 w0 = sb[i][0], w1 = sb[i][1];
    unsigned int* b = (unsigned int*)Bs;          // uint pitch = AST/2 = 20
    b[(nq + 0) * 20 + kk2] = f2bf_pk(w0.x, w1.x);
    b[(nq + 1) * 20 + kk2] = f2bf_pk(w0.y, w1.y);
    b[(nq + 2) * 20 + kk2] = f2bf_pk(w0.z, w1.z);
    b[(nq + 3) * 20 + kk2] = f2bf_pk(w0.w, w1.w);
  }
}

__global__ __launch_bounds__(256) void gemm1_kernel(const float* __restrict__ X,
                                                    const float* __restrict__ W1,
                                                    float* __restrict__ H1) {
  __shared__ unsigned short As[2][128 * AST];
  __shared__ unsigned short Bs[2][128 * AST];
  const int t = threadIdx.x, wave = t >> 5, lane = t & 31;
  const int m0 = blockIdx.x * 128;
  const int KT = (F_IN + 31) / 32;      // 273 tiles

  float2 sa[4][2];
  float4 sb[2][2];
  v8f acc[8] = {};

  g1_fetch(X, W1, m0, 0, t, sa, sb);
  g1_stage(As[0], Bs[0], t, sa, sb);
  __syncthreads();

  const int arow = (wave << 4) + (lane & 15);
  const int kbA  = (lane >> 4) << 3;
  const int kbB  = (lane >> 4) << 3;

  for (int kt = 0; kt < KT; ++kt) {
    if (kt + 1 < KT) g1_fetch(X, W1, m0, (kt + 1) << 5, t, sa, sb);  // overlap w/ WMMA

    // preload entire fragment set first so ds_loads pipeline ahead of the WMMA chain
    const unsigned short* Ab = &As[kt & 1][arow * AST];
    const unsigned short* Bb = &Bs[kt & 1][(lane & 15) * AST];
    v16bf aF = frag_a(Ab, kbA);
    v16bf bF[8];
#pragma unroll
    for (int nt = 0; nt < 8; ++nt) bF[nt] = frag_b(Bb + (nt << 4) * AST, kbB);
#pragma unroll
    for (int nt = 0; nt < 8; ++nt)
      acc[nt] = __builtin_amdgcn_wmma_f32_16x16x32_bf16(
          false, aF, false, bF[nt], (short)0, acc[nt], false, false);

    __syncthreads();
    if (kt + 1 < KT) g1_stage(As[(kt + 1) & 1], Bs[(kt + 1) & 1], t, sa, sb);
    __syncthreads();
  }

  const int mbase = m0 + (wave << 4) + ((lane >> 4) << 3);
  const int nbase = lane & 15;
#pragma unroll
  for (int nt = 0; nt < 8; ++nt)
#pragma unroll
    for (int r = 0; r < 8; ++r) {
      int row = mbase + r;
      if (row < N_NODES) H1[(size_t)row * HID + (nt << 4) + nbase] = acc[nt][r];
    }
}

// ---------------- GEMM2: Z2[N,80] = Hact[N,128] @ W2pad[128,80] --------------
#define BST2 136   // LDS pitch (272B, 16B aligned)

__global__ __launch_bounds__(256) void gemm2_kernel(const float* __restrict__ Hact,
                                                    const float* __restrict__ W2,
                                                    float* __restrict__ Z2) {
  __shared__ unsigned short Bs[CPAD * BST2];
  const int t = threadIdx.x, wave = t >> 5, lane = t & 31;
  const int m0 = blockIdx.x * 128;

  // stage W2 transposed + zero-padded; convert k-pairs, packed dword LDS stores
  for (int idx = t; idx < CPAD * (HID / 2); idx += 256) {   // 5120 pairs
    int n  = idx >> 6;          // 0..79
    int k2 = idx & 63;
    int k  = k2 << 1;
    float v0 = (n < NCLS) ? W2[(size_t)k * NCLS + n] : 0.f;
    float v1 = (n < NCLS) ? W2[(size_t)(k + 1) * NCLS + n] : 0.f;
    ((unsigned int*)Bs)[n * (BST2 / 2) + k2] = f2bf_pk(v0, v1);  // BST2/2 = 68
  }
  __syncthreads();

  v8f acc[5] = {};
  const int row  = m0 + (wave << 4) + (lane & 15);
  const int kbA  = (lane >> 4) << 3;
  const int kbB  = (lane >> 4) << 3;
  const float* prow = Hact + (size_t)row * HID;

#pragma unroll
  for (int ks = 0; ks < 4; ++ks) {
    int k0 = ks << 5;
    union { unsigned int u[8]; v16bf v; } af;
    if (row < N_NODES) {                               // wave-uniform (16-row strips)
      const float* p = prow + k0 + kbA;
      float4 a0 = *(const float4*)(p);
      float4 a1 = *(const float4*)(p + 4);
      float4 b0 = *(const float4*)(p + 16);
      float4 b1 = *(const float4*)(p + 20);
      af.u[0] = f2bf_pk(a0.x, a0.y);  af.u[1] = f2bf_pk(a0.z, a0.w);
      af.u[2] = f2bf_pk(a1.x, a1.y);  af.u[3] = f2bf_pk(a1.z, a1.w);
      af.u[4] = f2bf_pk(b0.x, b0.y);  af.u[5] = f2bf_pk(b0.z, b0.w);
      af.u[6] = f2bf_pk(b1.x, b1.y);  af.u[7] = f2bf_pk(b1.z, b1.w);
    } else {
#pragma unroll
      for (int j = 0; j < 8; ++j) af.u[j] = 0u;
    }
    const unsigned short* Bb = &Bs[(lane & 15) * BST2 + k0];
    v16bf bF[5];
#pragma unroll
    for (int nt = 0; nt < 5; ++nt) bF[nt] = frag_b(Bb + (nt << 4) * BST2, kbB);
#pragma unroll
    for (int nt = 0; nt < 5; ++nt)
      acc[nt] = __builtin_amdgcn_wmma_f32_16x16x32_bf16(
          false, af.v, false, bF[nt], (short)0, acc[nt], false, false);
  }

  const int mb = m0 + (wave << 4) + ((lane >> 4) << 3);
#pragma unroll
  for (int nt = 0; nt < 5; ++nt)
#pragma unroll
    for (int r = 0; r < 8; ++r) {
      int row2 = mb + r;
      if (row2 < N_NODES) Z2[(size_t)row2 * CPAD + (nt << 4) + (lane & 15)] = acc[nt][r];
    }
}

// ---------------- graph aggregation -----------------------------------------
__global__ void deg_kernel(const int* __restrict__ dst, float* __restrict__ deg) {
  int e = blockIdx.x * blockDim.x + threadIdx.x;
  if (e < NEDGE) atomicAdd(&deg[dst[e]], 1.0f);
}

__global__ void dinv_kernel(float* __restrict__ deg) {
  int i = blockIdx.x * blockDim.x + threadIdx.x;
  if (i < N_NODES) deg[i] = rsqrtf(deg[i] + 1.0f);   // +1 self-loop
}

__global__ void agg1_kernel(const int* __restrict__ src, const int* __restrict__ dst,
                            const float* __restrict__ dinv, const float* __restrict__ h1,
                            float* __restrict__ out) {
  int w = (blockIdx.x * blockDim.x + threadIdx.x) >> 5;   // wave per edge
  int lane = threadIdx.x & 31;
  if (w >= NEDGE) return;
  int s = src[w], d = dst[w];
  float nrm = dinv[s] * dinv[d];
  float4 v = ((const float4*)(h1 + (size_t)s * HID))[lane];
  float* op = out + (size_t)d * HID + (lane << 2);
  atomicAdd(op + 0, v.x * nrm);
  atomicAdd(op + 1, v.y * nrm);
  atomicAdd(op + 2, v.z * nrm);
  atomicAdd(op + 3, v.w * nrm);
}

__global__ void fin1_kernel(const float* __restrict__ h1, const float* __restrict__ dinv,
                            const float* __restrict__ b1, float* __restrict__ agg) {
  int idx = blockIdx.x * blockDim.x + threadIdx.x;        // N*128 threads
  int rowi = idx >> 7, f = idx & 127;
  float di = dinv[rowi];
  float v = agg[idx] + h1[idx] * di * di + b1[f];          // self-loop + bias
  agg[idx] = fmaxf(v, 0.f);                                // ReLU in place
}

__global__ void agg2_kernel(const int* __restrict__ src, const int* __restrict__ dst,
                            const float* __restrict__ dinv, const float* __restrict__ z2,
                            float* __restrict__ out2) {
  int w = (blockIdx.x * blockDim.x + threadIdx.x) >> 5;
  int lane = threadIdx.x & 31;
  if (w >= NEDGE) return;
  int s = src[w], d = dst[w];
  float nrm = dinv[s] * dinv[d];
  const float* zp = z2 + (size_t)s * CPAD;
  float* op = out2 + (size_t)d * CPAD;
  atomicAdd(op + lane,      zp[lane]      * nrm);
  atomicAdd(op + lane + 32, zp[lane + 32] * nrm);
  if (lane < NCLS - 64) atomicAdd(op + lane + 64, zp[lane + 64] * nrm);
}

__global__ void softmax_kernel(const float* __restrict__ z2, const float* __restrict__ out2,
                               const float* __restrict__ dinv, const float* __restrict__ b2,
                               float* __restrict__ out) {
  int w = (blockIdx.x * blockDim.x + threadIdx.x) >> 5;   // wave per row
  int lane = threadIdx.x & 31;
  if (w >= N_NODES) return;
  float di = dinv[w], sn = di * di;
  const float* zp = z2 + (size_t)w * CPAD;
  const float* ap = out2 + (size_t)w * CPAD;
  float v0 = ap[lane]      + zp[lane]      * sn + b2[lane];
  float v1 = ap[lane + 32] + zp[lane + 32] * sn + b2[lane + 32];
  float v2 = (lane < 6) ? (ap[lane + 64] + zp[lane + 64] * sn + b2[lane + 64]) : -1e30f;
  float m = fmaxf(v0, fmaxf(v1, v2));
#pragma unroll
  for (int o = 16; o > 0; o >>= 1) m = fmaxf(m, __shfl_xor(m, o, 32));
  float e0 = __expf(v0 - m), e1 = __expf(v1 - m);
  float e2 = (lane < 6) ? __expf(v2 - m) : 0.f;
  float s = e0 + e1 + e2;
#pragma unroll
  for (int o = 16; o > 0; o >>= 1) s += __shfl_xor(s, o, 32);
  float inv = 1.0f / s;
  float* po = out + (size_t)w * NCLS;
  po[lane]      = e0 * inv;
  po[lane + 32] = e1 * inv;
  if (lane < 6) po[lane + 64] = e2 * inv;
}

// ---------------- launch -----------------------------------------------------
extern "C" void kernel_launch(void* const* d_in, const int* in_sizes, int n_in,
                              void* d_out, int out_size, void* d_ws, size_t ws_size,
                              hipStream_t stream) {
  const float* x    = (const float*)d_in[0];
  const int*   edges= (const int*)d_in[1];
  const float* W1   = (const float*)d_in[2];
  const float* b1   = (const float*)d_in[3];
  const float* W2   = (const float*)d_in[4];
  const float* b2   = (const float*)d_in[5];
  float* out = (float*)d_out;
  float* ws  = (float*)d_ws;

  const int* srcp = edges;
  const int* dstp = edges + NEDGE;

  // workspace layout (floats): deg | h1 | agg1/hact | z2 | out2  (~33.4 MB)
  float* deg  = ws;                                   // 20000
  float* h1   = ws + 20224;                           // N*128
  float* agg1 = h1   + (size_t)N_NODES * HID;         // N*128
  float* z2   = agg1 + (size_t)N_NODES * HID;         // N*80
  float* out2 = z2   + (size_t)N_NODES * CPAD;        // N*80

  hipMemsetAsync(deg,  0, (size_t)N_NODES * sizeof(float), stream);
  hipMemsetAsync(agg1, 0, (size_t)N_NODES * HID * sizeof(float), stream);
  hipMemsetAsync(out2, 0, (size_t)N_NODES * CPAD * sizeof(float), stream);

  deg_kernel    <<<(NEDGE + 255) / 256, 256, 0, stream>>>(dstp, deg);
  dinv_kernel   <<<(N_NODES + 255) / 256, 256, 0, stream>>>(deg);
  gemm1_kernel  <<<(N_NODES + 127) / 128, 256, 0, stream>>>(x, W1, h1);
  agg1_kernel   <<<NEDGE / 8, 256, 0, stream>>>(srcp, dstp, deg, h1, agg1);
  fin1_kernel   <<<(N_NODES * HID) / 256, 256, 0, stream>>>(h1, deg, b1, agg1);
  gemm2_kernel  <<<(N_NODES + 127) / 128, 256, 0, stream>>>(agg1, W2, z2);
  agg2_kernel   <<<NEDGE / 8, 256, 0, stream>>>(srcp, dstp, deg, z2, out2);
  softmax_kernel<<<(N_NODES + 7) / 8, 256, 0, stream>>>(z2, out2, deg, b2, out);
}
